// CausalSelfAttention_77146202571225
// MI455X (gfx1250) — matrix operand order repaired and checked by reading, not compiled, not run on hardware
//
#include <hip/hip_runtime.h>

// ---------------------------------------------------------------------------
// Causal self-attention for MI455X (gfx1250), bf16 WMMA with fp32 accumulate.
// B=2, T=2048, D=1024, H=16, HD=64.
// Round 2: GEMM A-panels staged in LDS via GLOBAL_LOAD_ASYNC_TO_LDS_B128
// (ASYNCcnt-tracked, double-buffered), 2 M-tiles per wave for B reuse.
// ---------------------------------------------------------------------------

typedef __attribute__((ext_vector_type(16))) __bf16 bf16x16;
typedef __attribute__((ext_vector_type(8)))  __bf16 bf16x8;
typedef __attribute__((ext_vector_type(8)))  float  f32x8;

#define CAT16(lo, hi) __builtin_shufflevector((lo), (hi), 0,1,2,3,4,5,6,7,8,9,10,11,12,13,14,15)
#define WMMA_BF16(a, b, c) \
  __builtin_amdgcn_wmma_f32_16x16x32_bf16(false, (a), false, (b), (short)0, (c), false, false)

// fp32 -> bf16 round-to-nearest-even (storage only; no bf16 arithmetic).
static __device__ __forceinline__ __bf16 f2bf(float f) {
  union { float f; unsigned u; } in; in.f = f;
  unsigned u = in.u;
  unsigned r = (u + 0x7FFFu + ((u >> 16) & 1u)) >> 16;
  union { unsigned short s; __bf16 b; } out; out.s = (unsigned short)r;
  return out.b;
}

static __device__ __forceinline__ float rowmax16(float v) {
  v = fmaxf(v, __shfl_xor(v, 1, 32));
  v = fmaxf(v, __shfl_xor(v, 2, 32));
  v = fmaxf(v, __shfl_xor(v, 4, 32));
  v = fmaxf(v, __shfl_xor(v, 8, 32));
  return v;
}
static __device__ __forceinline__ float rowsum16(float v) {
  v += __shfl_xor(v, 1, 32);
  v += __shfl_xor(v, 2, 32);
  v += __shfl_xor(v, 4, 32);
  v += __shfl_xor(v, 8, 32);
  return v;
}

// Per-lane async copy of 16B global -> LDS, tracked by ASYNCcnt.
static __device__ __forceinline__ void async_g2l_b128(unsigned ldsOff,
                                                      unsigned long long gaddr) {
  asm volatile("global_load_async_to_lds_b128 %0, %1, off"
               :: "v"(ldsOff), "v"(gaddr) : "memory");
}
static __device__ __forceinline__ void wait_asynccnt0() {
  asm volatile("s_wait_asynccnt 0x0" ::: "memory");
}

// ---------------------------------------------------------------------------
// fp32 -> bf16 conversion, 4 elements/thread.
// ---------------------------------------------------------------------------
__global__ void cvt_f32_bf16(const float* __restrict__ in, __bf16* __restrict__ out, int n) {
  int i = (blockIdx.x * blockDim.x + threadIdx.x) * 4;
  if (i + 3 < n) {
    float4 v = *(const float4*)(in + i);
    out[i + 0] = f2bf(v.x);
    out[i + 1] = f2bf(v.y);
    out[i + 2] = f2bf(v.z);
    out[i + 3] = f2bf(v.w);
  }
}

// ---------------------------------------------------------------------------
// GEMM: out[m,n] = sum_k A[m,k] * W[n,k] + bias[n]
// A row-major [M,K] bf16; W row-major [N,K] bf16 (torch Linear weight).
// Block = 128 threads (4 waves), tile 32M x 64N: wave w owns N-tile w,
// and computes TWO 16x16 M-tiles (B operand reused for 2 WMMAs).
// A panels (32 rows x 128 K, 8 KB) are staged in double-buffered LDS via
// GLOBAL_LOAD_ASYNC_TO_LDS_B128 by all 128 threads and shared by all waves:
//   issue(buf p) -> s_wait_asynccnt 0 -> barrier -> compute(buf p)
// A wave issues panel p+1 (other buffer) only after barrier p, by which time
// every wave has finished reading panel p-1 from that buffer -> race-free.
// Operand layouts (wave32 ISA):
//   A 16x32: lanes 0-15: K0-7 (v0-3), K16-23 (v4-7); lanes 16-31: K8-15,K24-31
//   B 32x16: lane = n%16; same K split as A (lanes 0-15: K0-7,K16-23; 16-31: K8-15,K24-31)
// ---------------------------------------------------------------------------
__global__ __launch_bounds__(128) void gemm_bf16_wmma(
    const __bf16* __restrict__ A, const __bf16* __restrict__ W,
    const float* __restrict__ bias, __bf16* __restrict__ outBf,
    float* __restrict__ outF, int N, int K)
{
  __shared__ __align__(16) __bf16 sA[2][32 * 128];   // 2 x 8 KB panels

  const int t     = threadIdx.x;
  const int lane  = t & 31;
  const int wv    = t >> 5;
  const int half  = lane >> 4;
  const int l16   = lane & 15;
  const int mBase = blockIdx.x * 32;
  const int nBase = blockIdx.y * 64 + wv * 16;

  const __bf16* bRow = W + (size_t)(nBase + l16) * K + half * 8;

  f32x8 c0 = {}, c1 = {};
  const int nPanels = K >> 7;                        // K / 128

  for (int p = 0; p < nPanels; ++p) {
    const int kp  = p << 7;
    const int buf = p & 1;

    // ---- stage A panel: 32 rows x 128 K = 512 b128 chunks, 4 per thread ----
    {
      const __bf16* gp = A + (size_t)mBase * K + kp;
      const unsigned ldsBase = (unsigned)(uintptr_t)&sA[buf][0];
#pragma unroll
      for (int i = 0; i < 4; ++i) {
        const unsigned chunk = (unsigned)t + i * 128; // 0..511
        const unsigned row   = chunk >> 4;            // 0..31
        const unsigned col   = (chunk & 15) * 8;      // bf16 elements
        async_g2l_b128(ldsBase + row * 256 + col * 2,
                       (unsigned long long)(uintptr_t)(gp + (size_t)row * K + col));
      }
      wait_asynccnt0();
      __syncthreads();
    }

    // ---- consume panel: 4 K-steps of 32 ----
#pragma unroll
    for (int k4 = 0; k4 < 4; ++k4) {
      const int k = k4 * 32;
      // B operand from global (contiguous 16B per lane)
      __builtin_prefetch(bRow + kp + k + 256, 0, 3);
      bf16x8 blo = *(const bf16x8*)(bRow + kp + k);
      bf16x8 bhi = *(const bf16x8*)(bRow + kp + k + 16);
      bf16x16 b  = CAT16(blo, bhi);
      // A operands for both M-tiles from LDS
      const __bf16* p0 = &sA[buf][(size_t)l16 * 128 + k + half * 8];
      const __bf16* p1 = &sA[buf][(size_t)(16 + l16) * 128 + k + half * 8];
      bf16x16 a0 = CAT16(*(const bf16x8*)p0, *(const bf16x8*)(p0 + 16));
      bf16x16 a1 = CAT16(*(const bf16x8*)p1, *(const bf16x8*)(p1 + 16));
      c0 = WMMA_BF16(a0, b, c0);
      c1 = WMMA_BF16(a1, b, c1);
    }
  }

  // ---- epilogue: C layout lane l, vgpr r -> row = r + 8*(l/16), col = l%16 ----
  const int col = nBase + l16;
  const float bv = bias ? bias[col] : 0.0f;
#pragma unroll
  for (int r = 0; r < 8; ++r) {
    const int row0 = mBase + r + 8 * half;
    const int row1 = row0 + 16;
    const float v0 = c0[r] + bv;
    const float v1 = c1[r] + bv;
    if (outBf) {
      outBf[(size_t)row0 * N + col] = f2bf(v0);
      outBf[(size_t)row1 * N + col] = f2bf(v1);
    } else {
      outF[(size_t)row0 * N + col] = v0;
      outF[(size_t)row1 * N + col] = v1;
    }
  }
}

// ---------------------------------------------------------------------------
// Flash attention (causal, online softmax).
// qkv: [B*T, 3*D] bf16 rows = {q | k | v}, head h at column h*64 (+D, +2D).
// One wave handles one 16-query tile of one (b,h); loops over 32-key blocks.
// S = Q*K^T: 4 WMMAs (K operand straight from global, contiguous in hd).
// P*V: P goes through per-wave LDS (C-layout -> A-layout), V staged
// transposed in LDS so the B operand reads are contiguous ds_load_b128.
// Same-wave LDS store->load is in-order per the ISA (DScnt), no barrier.
// ---------------------------------------------------------------------------
__global__ __launch_bounds__(128) void flash_attn_bf16(
    const __bf16* __restrict__ qkv, __bf16* __restrict__ outBf)
{
  constexpr int T = 2048, H = 16, HD = 64, D = 1024, C3 = 3072;
  __shared__ __align__(16) __bf16 sVt[4][HD * 32];  // [hd][key] transposed V tile
  __shared__ __align__(16) __bf16 sP [4][16 * 32];  // P tile, row-major

  const int lane = threadIdx.x & 31;
  const int wv   = threadIdx.x >> 5;
  const int half = lane >> 4;
  const int l16  = lane & 15;

  const int wid = blockIdx.x * 4 + wv;     // 4096 waves total
  const int qt  = wid & (T / 16 - 1);      // query tile
  const int h   = (wid >> 7) & (H - 1);
  const int b   = wid >> 11;
  const int qb  = qt * 16;

  const __bf16* qBase = qkv + (size_t)(b * T) * C3 + h * HD;
  const __bf16* kBase = qBase + D;
  const __bf16* vBase = qBase + 2 * D;

  // Q A-operands for hd 0..31 and 32..63 (loaded once).
  const __bf16* qRow = qBase + (size_t)(qb + l16) * C3 + half * 8;
  bf16x16 qa[2];
#pragma unroll
  for (int s = 0; s < 2; ++s) {
    bf16x8 lo = *(const bf16x8*)(qRow + s * 32);
    bf16x8 hi = *(const bf16x8*)(qRow + s * 32 + 16);
    qa[s] = CAT16(lo, hi);
  }

  float m[8], l[8];
  f32x8 o[4] = {{}, {}, {}, {}};
#pragma unroll
  for (int r = 0; r < 8; ++r) { m[r] = -1e30f; l[r] = 0.0f; }

  const float sc = 0.125f * 1.44269504f;       // 1/sqrt(64) folded with log2(e)
  const int nkb = (qb + 47) >> 5;              // ceil((qb+16)/32) key blocks

  for (int kb = 0; kb < nkb * 32; kb += 32) {
    // ---- stage V^T tile: sVt[hd][key], key = lane ----
    const __bf16* vRow = vBase + (size_t)(kb + lane) * C3;
#pragma unroll
    for (int j = 0; j < 8; ++j) {
      bf16x8 v8 = *(const bf16x8*)(vRow + j * 8);
#pragma unroll
      for (int e = 0; e < 8; ++e) sVt[wv][(j * 8 + e) * 32 + lane] = v8[e];
    }

    // ---- S = Q * K^T over two 16-key tiles ----
    f32x8 s[2] = {{}, {}};
#pragma unroll
    for (int t2 = 0; t2 < 2; ++t2) {
      const __bf16* kRow = kBase + (size_t)(kb + t2 * 16 + l16) * C3 + half * 8;
#pragma unroll
      for (int ks = 0; ks < 2; ++ks) {
        bf16x8 lo = *(const bf16x8*)(kRow + ks * 32);
        bf16x8 hi = *(const bf16x8*)(kRow + ks * 32 + 16);
        bf16x16 bm = CAT16(lo, hi);
        s[t2] = WMMA_BF16(qa[ks], bm, s[t2]);
      }
    }

    // ---- causal mask + online softmax update (per output row r) ----
#pragma unroll
    for (int r = 0; r < 8; ++r) {
      const int row = qb + r + 8 * half;
      float s0 = s[0][r] * sc;
      float s1 = s[1][r] * sc;
      if (kb + l16 > row)      s0 = -1e30f;
      if (kb + 16 + l16 > row) s1 = -1e30f;
      const float mx = rowmax16(fmaxf(s0, s1));
      const float mn = fmaxf(m[r], mx);
      const float al = __builtin_amdgcn_exp2f(m[r] - mn);
      const float p0 = __builtin_amdgcn_exp2f(s0 - mn);
      const float p1 = __builtin_amdgcn_exp2f(s1 - mn);
      l[r] = l[r] * al + rowsum16(p0 + p1);
      m[r] = mn;
#pragma unroll
      for (int nt = 0; nt < 4; ++nt) o[nt][r] *= al;
      sP[wv][(r + 8 * half) * 32 + l16]      = f2bf(p0);
      sP[wv][(r + 8 * half) * 32 + 16 + l16] = f2bf(p1);
    }

    // ---- O += P * V (K = 32 keys), 4 hd tiles ----
    {
      const __bf16* pRow = &sP[wv][l16 * 32 + half * 8];
      bf16x8 plo = *(const bf16x8*)(pRow);
      bf16x8 phi = *(const bf16x8*)(pRow + 16);
      bf16x16 pa = CAT16(plo, phi);
#pragma unroll
      for (int nt = 0; nt < 4; ++nt) {
        const __bf16* vtRow = &sVt[wv][(nt * 16 + l16) * 32 + half * 8];
        bf16x8 vlo = *(const bf16x8*)(vtRow);
        bf16x8 vhi = *(const bf16x8*)(vtRow + 16);
        bf16x16 vb = CAT16(vlo, vhi);
        o[nt] = WMMA_BF16(pa, vb, o[nt]);
      }
    }
  }

  // ---- epilogue: normalize and store bf16 [B*T, D] ----
#pragma unroll
  for (int r = 0; r < 8; ++r) {
    const int row = qb + r + 8 * half;
    const float inv = 1.0f / l[r];
#pragma unroll
    for (int nt = 0; nt < 4; ++nt) {
      const float v = o[nt][r] * inv;
      outBf[((size_t)(b * T) + row) * D + h * HD + nt * 16 + l16] = f2bf(v);
    }
  }
}

// ---------------------------------------------------------------------------
// Host-side launch chain.
// ---------------------------------------------------------------------------
extern "C" void kernel_launch(void* const* d_in, const int* in_sizes, int n_in,
                              void* d_out, int out_size, void* d_ws, size_t ws_size,
                              hipStream_t stream) {
  constexpr int B = 2, T = 2048, D = 1024, H = 16;
  constexpr int M = B * T;        // 4096 rows
  (void)in_sizes; (void)n_in; (void)out_size; (void)ws_size; (void)H;

  const float* x      = (const float*)d_in[0];
  const float* w_qkv  = (const float*)d_in[1];
  const float* b_qkv  = (const float*)d_in[2];
  const float* w_proj = (const float*)d_in[3];
  const float* b_proj = (const float*)d_in[4];
  float* out = (float*)d_out;

  char* ws = (char*)d_ws;
  size_t off = 0;
  auto carve = [&](size_t bytes) -> void* {
    void* p = ws + off;
    off = (off + bytes + 255) & ~(size_t)255;
    return p;
  };
  __bf16* xb     = (__bf16*)carve((size_t)M * D * 2);         //  8 MB
  __bf16* wqkvb  = (__bf16*)carve((size_t)3 * D * D * 2);     //  6 MB
  __bf16* wprojb = (__bf16*)carve((size_t)D * D * 2);         //  2 MB
  __bf16* qkvb   = (__bf16*)carve((size_t)M * 3 * D * 2);     // 24 MB
  __bf16* attnb  = (__bf16*)carve((size_t)M * D * 2);         //  8 MB

  // 1) fp32 -> bf16 conversions
  {
    int n = M * D;
    cvt_f32_bf16<<<(n / 4 + 255) / 256, 256, 0, stream>>>(x, xb, n);
    n = 3 * D * D;
    cvt_f32_bf16<<<(n / 4 + 255) / 256, 256, 0, stream>>>(w_qkv, wqkvb, n);
    n = D * D;
    cvt_f32_bf16<<<(n / 4 + 255) / 256, 256, 0, stream>>>(w_proj, wprojb, n);
  }

  // 2) QKV projection: [M,1024] x [3072,1024]^T -> [M,3072] bf16 (+bias)
  gemm_bf16_wmma<<<dim3(M / 32, (3 * D) / 64), 128, 0, stream>>>(
      xb, wqkvb, b_qkv, qkvb, nullptr, 3 * D, D);

  // 3) Flash attention: 4096 waves (one per 16-query tile per head), 4/block
  flash_attn_bf16<<<dim3(1024), 128, 0, stream>>>(qkvb, attnb);

  // 4) Output projection: [M,1024] x [1024,1024]^T -> fp32 d_out (+bias)
  gemm_bf16_wmma<<<dim3(M / 32, D / 64), 128, 0, stream>>>(
      attnb, wprojb, b_proj, nullptr, out, D, D);
}
